// _dconv_58901181497894
// MI455X (gfx1250) — matrix operand order repaired and checked by reading, not compiled, not run on hardware
//
#include <hip/hip_runtime.h>
#include <hip/hip_bf16.h>

typedef __attribute__((ext_vector_type(2)))  float    v2f;
typedef __attribute__((ext_vector_type(8)))  float    v8f;
typedef __attribute__((ext_vector_type(8)))  _Float16 v8h;
typedef __attribute__((ext_vector_type(16))) _Float16 v16h;

__device__ __forceinline__ v8f wmma_f32_k4(v2f a, v2f b, v8f c) {
  // D = A(16x4 f32) * B(4x16 f32) + C(16x16 f32)
  return __builtin_amdgcn_wmma_f32_16x16x4_f32(
      false, a, false, b, (short)0, c, false, false);
}

__device__ __forceinline__ v8f wmma_f32_f16_k32(v16h a, v16h b, v8f c) {
  // D = A(16x32 f16) * B(32x16 f16) + C(16x16 f32)
  return __builtin_amdgcn_wmma_f32_16x16x32_f16(
      false, a, false, b, (short)0, c, false, false);
}

__device__ __forceinline__ float sigm(float x) {
  return 1.0f / (1.0f + __expf(-x));
}

// ---------------------------------------------------------------------------
// Convert U (256x1024 f32 row-major, K x N) into Ut (1024x256 f16, N x K):
// lets each lane's f16-WMMA B fragment be contiguous 16-half chunks.
// ---------------------------------------------------------------------------
__global__ __launch_bounds__(256)
void convert_Ut(const float* __restrict__ U, _Float16* __restrict__ Ut) {
  const int idx = blockIdx.x * 256 + threadIdx.x;   // 262144 total
  const int k = idx >> 10, n = idx & 1023;
  Ut[n * 256 + k] = (_Float16)U[k * 1024 + n];
}

// ---------------------------------------------------------------------------
// Kernel 1: xw_enc[m][n] = sum_k X[bi][ti][k] * W_enc[k][n], m = ti*4+bi
// M=1024, K=256, N=1024.  One wave per 16x16 tile, f32 WMMA (full precision).
// ---------------------------------------------------------------------------
__global__ __launch_bounds__(32)
void gemm_xw_enc(const float* __restrict__ X, const float* __restrict__ W,
                 float* __restrict__ out) {
  const int lane = threadIdx.x;
  const int lm   = lane & 15;
  const int koff = (lane < 16) ? 0 : 2;
  const int n0 = blockIdx.x * 16;
  const int m0 = blockIdx.y * 16;

  const int row = m0 + lm;
  const int bi = row & 3, ti = row >> 2;
  const float* arow = X + bi * 65536 + ti * 256;

  v8f acc = {};
  for (int k = 0; k < 256; k += 4) {
    v2f a, b;
    a.x = arow[k + koff];
    a.y = arow[k + koff + 1];
    b.x = W[(k + koff) * 1024 + n0 + lm];
    b.y = W[(k + koff + 1) * 1024 + n0 + lm];
    acc = wmma_f32_k4(a, b, acc);
  }
  const int rbase = (lane < 16) ? 0 : 8;
#pragma unroll
  for (int r = 0; r < 8; ++r)
    out[(m0 + rbase + r) * 1024 + n0 + lm] = acc[r];
}

// ---------------------------------------------------------------------------
// Kernel 2: xw_dec[m][n], m = s*4+bi, A-row = concat(x[s], x[255-s])
// M=1020 (padded tiles), K=512, N=1024, f32 WMMA.
// ---------------------------------------------------------------------------
__global__ __launch_bounds__(32)
void gemm_xw_dec(const float* __restrict__ X, const float* __restrict__ W,
                 float* __restrict__ out) {
  const int lane = threadIdx.x;
  const int lm   = lane & 15;
  const int koff = (lane < 16) ? 0 : 2;
  const int n0 = blockIdx.x * 16;
  const int m0 = blockIdx.y * 16;

  int row = m0 + lm;
  int rowc = row < 1019 ? row : 1019;   // clamp pad rows (stores are guarded)
  const int s = rowc >> 2, bi = rowc & 3;
  const float* fwdrow = X + bi * 65536 + s * 256;
  const float* bwdrow = X + bi * 65536 + (255 - s) * 256;

  v8f acc = {};
  for (int k = 0; k < 512; k += 4) {
    const int ka = k + koff;            // even, so ka/ka+1 in same half
    const float* rp = (ka < 256) ? (fwdrow + ka) : (bwdrow + (ka - 256));
    v2f a, b;
    a.x = rp[0];
    a.y = rp[1];
    b.x = W[ka * 1024 + n0 + lm];
    b.y = W[(ka + 1) * 1024 + n0 + lm];
    acc = wmma_f32_k4(a, b, acc);
  }
  const int rbase = (lane < 16) ? 0 : 8;
#pragma unroll
  for (int r = 0; r < 8; ++r) {
    const int orow = m0 + rbase + r;
    if (orow < 1020) out[orow * 1024 + n0 + lm] = acc[r];
  }
}

// ---------------------------------------------------------------------------
// Kernel 3: persistent single-workgroup LSTM scan.
// Recurrent h@U in f16 WMMA 16x16x32 (f32 accumulate): 512 WMMA/step vs 4096
// for the f32 K=4 path.  h kept in LDS as f16 (A fragments = contiguous b128
// DS loads, K 16h..16h+15 per lane half, same order as B); Ut is f16 N-major so B fragments are
// contiguous b128 global loads (L2-resident, 512KB).
// Gates / c / outputs stay fp32.
// ---------------------------------------------------------------------------
__global__ __launch_bounds__(1024)
void lstm_scan_kernel(const float* __restrict__ xw, const _Float16* __restrict__ Ut,
                      const float* __restrict__ bias, const float* __restrict__ h0,
                      const float* __restrict__ c0, float* __restrict__ Fout,
                      float* __restrict__ hT, float* __restrict__ cT, int T) {
  __shared__ _Float16 hbuf[16 * 256];   // padded f16 h (rows 4..15 stay zero)
  __shared__ float    cbuf[1024];       // c (4 x 256) fp32
  __shared__ float    zbuf[4 * 1024];   // z = h @ U, fp32

  const int tid  = threadIdx.x;
  const int lane = tid & 31;
  const int wave = tid >> 5;
  const int lm   = lane & 15;
  const int aoff = (lane < 16) ? 0 : 16;   // f16 A: lanes 0-15 K 0..15
  const int boff = (lane < 16) ? 0 : 16;   // f16 B: lanes 0-15 K 0..15
  const int bi = tid >> 8, j = tid & 255;

  for (int idx = tid; idx < 16 * 256; idx += 1024) {
    const int row = idx >> 8;
    hbuf[idx] = (row < 4 && h0) ? (_Float16)h0[idx] : (_Float16)0.0f;
  }
  cbuf[tid] = c0 ? c0[tid] : 0.0f;
  __syncthreads();

  for (int step = 0; step < T; ++step) {
    // z = h @ U : each wave computes 2 of the 64 N-tiles (M=16pad,K=256)
#pragma unroll
    for (int tn = 0; tn < 2; ++tn) {
      const int n0 = (wave * 2 + tn) * 16;
      const _Float16* bcol = Ut + (n0 + lm) * 256 + boff;
      const _Float16* arow = hbuf + lm * 256 + aoff;
      v8f acc = {};
#pragma unroll
      for (int kb = 0; kb < 256; kb += 32) {
        v16h a = *(const v16h*)(arow + kb);        // K kb+aoff .. +15
        v16h b = *(const v16h*)(bcol + kb);        // K kb+boff .. +15, col n
        acc = wmma_f32_f16_k32(a, b, acc);
      }
      if (lane < 16) {
#pragma unroll
        for (int r = 0; r < 4; ++r)                // batch rows 0..3 are real
          zbuf[r * 1024 + n0 + lm] = acc[r];
      }
    }
    __syncthreads();

    // Gate math (fp32): one thread per (bi, j) state element
    const float* xws = xw + step * 4096 + bi * 1024;
    const float zi = zbuf[bi * 1024 + j]       + xws[j]       + bias[j];
    const float zf = zbuf[bi * 1024 + 256 + j] + xws[256 + j] + bias[256 + j];
    const float zg = zbuf[bi * 1024 + 512 + j] + xws[512 + j] + bias[512 + j];
    const float zo = zbuf[bi * 1024 + 768 + j] + xws[768 + j] + bias[768 + j];
    const float c = sigm(zf) * cbuf[tid] + sigm(zi) * tanhf(zg);
    const float h = sigm(zo) * tanhf(c);
    cbuf[tid] = c;
    hbuf[tid] = (_Float16)h;                         // rows 0..3 of padded h
    if (Fout) Fout[(step * 4 + bi) * 256 + j] = h;   // F rows m = step*4+bi
    __syncthreads();
  }
  if (hT) hT[tid] = (float)hbuf[tid];  // decoder consumes h only via f16 GEMM
  if (cT) cT[tid] = cbuf[tid];
}

// ---------------------------------------------------------------------------
// Kernel 4: four gated projections of F (1020x256) vs 256x256 weights (f32
// WMMA, 4 accumulators per wave); fused tanh/sigmoid + flip-scatter into D.
// D layout: (bi, 511, 256); rows [0..254]=flip(R), 255=mid, [256..510]=F1.
// ---------------------------------------------------------------------------
__global__ __launch_bounds__(32)
void post_kernel(const float* __restrict__ F,
                 const float* __restrict__ Wrg, const float* __restrict__ brg,
                 const float* __restrict__ Wfg, const float* __restrict__ bfg,
                 const float* __restrict__ Wrl, const float* __restrict__ brl,
                 const float* __restrict__ Wfl, const float* __restrict__ bfl,
                 float* __restrict__ D) {
  const int lane = threadIdx.x;
  const int lm   = lane & 15;
  const int koff = (lane < 16) ? 0 : 2;
  const int n0 = blockIdx.x * 16;   // N=256 -> 16 tiles
  const int m0 = blockIdx.y * 16;   // M=1020 padded -> 64 tiles

  int row = m0 + lm;
  int rowc = row < 1019 ? row : 1019;
  const float* arow = F + rowc * 256;

  v8f aRL = {}, aRG = {}, aFL = {}, aFG = {};
  for (int k = 0; k < 256; k += 4) {
    const int ka = k + koff;
    v2f a, b;
    a.x = arow[ka];
    a.y = arow[ka + 1];
    b.x = Wrl[ka * 256 + n0 + lm]; b.y = Wrl[(ka + 1) * 256 + n0 + lm];
    aRL = wmma_f32_k4(a, b, aRL);
    b.x = Wrg[ka * 256 + n0 + lm]; b.y = Wrg[(ka + 1) * 256 + n0 + lm];
    aRG = wmma_f32_k4(a, b, aRG);
    b.x = Wfl[ka * 256 + n0 + lm]; b.y = Wfl[(ka + 1) * 256 + n0 + lm];
    aFL = wmma_f32_k4(a, b, aFL);
    b.x = Wfg[ka * 256 + n0 + lm]; b.y = Wfg[(ka + 1) * 256 + n0 + lm];
    aFG = wmma_f32_k4(a, b, aFG);
  }

  const int col = n0 + lm;
  const int rbase = (lane < 16) ? 0 : 8;
#pragma unroll
  for (int r = 0; r < 8; ++r) {
    const int orow = m0 + rbase + r;
    if (orow < 1020) {
      const int s = orow >> 2, bb = orow & 3;
      const float f1 = tanhf(aFL[r] + bfl[col]) * sigm(aFG[r] + bfg[col]);
      const float rr = tanhf(aRL[r] + brl[col]) * sigm(aRG[r] + brg[col]);
      D[(bb * 511 + 256 + s) * 256 + col] = f1;   // F1 block
      D[(bb * 511 + 254 - s) * 256 + col] = rr;   // flipped R block
    }
  }
}

// Middle row of D: tanh(m) broadcast
__global__ __launch_bounds__(1024)
void mid_kernel(const float* __restrict__ m, float* __restrict__ D) {
  const int tid = threadIdx.x;
  const int bi = tid >> 8, col = tid & 255;
  D[(bi * 511 + 255) * 256 + col] = tanhf(m[col]);
}

// ---------------------------------------------------------------------------
// Kernel 5: banded gather, out[bi][i][j][:] = D[bi][255-i+j][:]
// One 64-thread block streams one 1KB row via float4 (HBM-bandwidth bound;
// D working set is 2MB -> reads come from L2).
// ---------------------------------------------------------------------------
__global__ __launch_bounds__(64)
void gather_kernel(const float* __restrict__ D, float* __restrict__ out) {
  const unsigned blk = blockIdx.x;           // bi*65536 + i*256 + j
  const int bi = blk >> 16;
  const int i  = (blk >> 8) & 255;
  const int jj = blk & 255;
  const int drow = 255 - i + jj;             // in [0, 510]
  const float4* src = (const float4*)(D + (unsigned)(bi * 511 + drow) * 256);
  float4* dst = (float4*)(out + (unsigned long long)blk * 256);
  dst[threadIdx.x] = src[threadIdx.x];
}

// ---------------------------------------------------------------------------
extern "C" void kernel_launch(void* const* d_in, const int* in_sizes, int n_in,
                              void* d_out, int out_size, void* d_ws, size_t ws_size,
                              hipStream_t stream) {
  const float* X     = (const float*)d_in[0];
  const float* W_enc = (const float*)d_in[1];
  const float* U_enc = (const float*)d_in[2];
  const float* b_enc = (const float*)d_in[3];
  const float* W_dec = (const float*)d_in[4];
  const float* U_dec = (const float*)d_in[5];
  const float* b_dec = (const float*)d_in[6];
  const float* W_rg  = (const float*)d_in[7];
  const float* b_rg  = (const float*)d_in[8];
  const float* W_fg  = (const float*)d_in[9];
  const float* b_fg  = (const float*)d_in[10];
  const float* W_rl  = (const float*)d_in[11];
  const float* b_rl  = (const float*)d_in[12];
  const float* W_fl  = (const float*)d_in[13];
  const float* b_fl  = (const float*)d_in[14];
  const float* m     = (const float*)d_in[15];

  float* ws     = (float*)d_ws;
  float* xw_enc = ws;                 // 1,048,576 floats (t*b x 4h)
  float* xw_dec = ws + 1048576;       // 1,048,576 floats (uses 1,044,480)
  float* hT     = ws + 2097152;       // 1024
  float* cT     = hT + 1024;          // 1024
  float* F      = cT + 1024;          // 261,120 (padded slot 262,144)
  float* D      = F + 262144;         // 523,264
  _Float16* Ut_enc = (_Float16*)(ws + 2884608);   // 262,144 halves (512KB)
  _Float16* Ut_dec = (_Float16*)(ws + 3015680);   // 262,144 halves (512KB)
  // total ~12.6 MB of workspace

  // 0) f16 transposed recurrent weights (L2-resident)
  convert_Ut<<<1024, 256, 0, stream>>>(U_enc, Ut_enc);
  convert_Ut<<<1024, 256, 0, stream>>>(U_dec, Ut_dec);
  // 1) parallel fp32 input projections (f32 WMMA GEMMs)
  gemm_xw_enc<<<dim3(64, 64), 32, 0, stream>>>(X, W_enc, xw_enc);
  gemm_xw_dec<<<dim3(64, 64), 32, 0, stream>>>(X, W_dec, xw_dec);
  // 2) sequential encoder scan (single WGP, state in LDS, f16 WMMA recurrence)
  lstm_scan_kernel<<<1, 1024, 0, stream>>>(xw_enc, Ut_enc, b_enc,
                                           nullptr, nullptr, nullptr, hT, cT, 256);
  // 3) sequential decoder scan, seeded with encoder final state
  lstm_scan_kernel<<<1, 1024, 0, stream>>>(xw_dec, Ut_dec, b_dec,
                                           hT, cT, F, nullptr, nullptr, 255);
  // 4) gated projections -> D (with flip), plus middle row
  post_kernel<<<dim3(16, 64), 32, 0, stream>>>(F, W_rg, b_rg, W_fg, b_fg,
                                               W_rl, b_rl, W_fl, b_fl, D);
  mid_kernel<<<1, 1024, 0, stream>>>(m, D);
  // 5) 256MB banded-gather stream to output
  gather_kernel<<<262144, 64, 0, stream>>>(D, (float*)d_out);
}
